// DecoderEncoderMultiHeadAttention_61993557951027
// MI455X (gfx1250) — matrix operand order repaired and checked by reading, not compiled, not run on hardware
//
#include <hip/hip_runtime.h>
#include <hip/hip_bf16.h>
#include <math.h>

// ---------------------------------------------------------------------------
// DecoderEncoderMultiHeadAttention for MI455X (gfx1250, wave32, WMMA)
// B=16, T_dec=T_enc=1024, D_MODEL=64, N_HEAD=4, HEAD_DIM=16
//
// ~4.8 GFLOP total vs ~12 MB min HBM traffic -> compute bound. All four
// matmuls (QKV proj, Q@K^T, P@V, out proj) run on v_wmma_f32_16x16x32_f16.
// ---------------------------------------------------------------------------

typedef __attribute__((ext_vector_type(16))) _Float16 v16h;
typedef __attribute__((ext_vector_type(8)))  _Float16 v8h;
typedef __attribute__((ext_vector_type(2)))  _Float16 v2h;
typedef __attribute__((ext_vector_type(8)))  float    v8f;

union Vec16 { v16h v; v8h h[2]; };

__device__ __forceinline__ v16h pack16(v8h lo, v8h hi) {
    Vec16 u; u.h[0] = lo; u.h[1] = hi; return u.v;
}

__device__ __forceinline__ v8f wmma_f16(v16h a, v16h b, v8f c) {
    // D = A(16x32 f16) * B(32x16 f16) + C(16x16 f32)
    return __builtin_amdgcn_wmma_f32_16x16x32_f16(
        /*neg_a=*/false, a, /*neg_b=*/false, b,
        /*c_mod=*/(short)0, c, /*reuse_a=*/false, /*reuse_b=*/false);
}

__device__ __forceinline__ float wave_max(float v) {
    #pragma unroll
    for (int m = 16; m >= 1; m >>= 1) v = fmaxf(v, __shfl_xor(v, m, 32));
    return v;
}
__device__ __forceinline__ float wave_sum(float v) {
    #pragma unroll
    for (int m = 16; m >= 1; m >>= 1) v += __shfl_xor(v, m, 32);
    return v;
}

// ---------------------------------------------------------------------------
// Kernel 0: transpose all weights to f16 B-fragment-friendly layout [n][k].
// grid = 256, block = 64.
// ---------------------------------------------------------------------------
__global__ void prep_w(const float* __restrict__ Wq, const float* __restrict__ Wkv,
                       const float* __restrict__ Wp,
                       _Float16* __restrict__ WqT, _Float16* __restrict__ WkvT,
                       _Float16* __restrict__ WpT)
{
    const int n = blockIdx.x, k = threadIdx.x;
    if (n < 64)       WqT [n * 64 + k]         = (_Float16)Wq [k * 64  + n];
    else if (n < 192) WkvT[(n - 64) * 64 + k]  = (_Float16)Wkv[k * 128 + (n - 64)];
    else              WpT [(n - 192) * 64 + k] = (_Float16)Wp [k * 64  + (n - 192)];
}

// ---------------------------------------------------------------------------
// Kernel 1: QKV projection as WMMA GEMM. One block per 16-row tile of the
// 16384 (b,t) rows; 4 waves, wave w = head w (16 output cols each of q/k/v).
// 6 WMMAs per wave. Bias + 1/sqrt(16) fused into the epilogue.
//   Qh[b][h][t][hd] f16   Kh[b][h][t][hd] f16   Vt[b][h][hd][t] f16
// grid = 1024, block = 128
// ---------------------------------------------------------------------------
__global__ void prep_qkv(const float* __restrict__ x, const float* __restrict__ enc,
                         const _Float16* __restrict__ WqT, const _Float16* __restrict__ WkvT,
                         const float* __restrict__ bkv, const float* __restrict__ bq,
                         _Float16* __restrict__ Qh, _Float16* __restrict__ Kh,
                         _Float16* __restrict__ Vt)
{
    const int rowbase = blockIdx.x * 16;          // aligned within one batch
    const int b = rowbase >> 10, t0 = rowbase & 1023;
    const int tid = threadIdx.x, w = tid >> 5, l = tid & 31;
    const int lh = l >> 4, ln = l & 15;

    // A fragments (16x32 f16) for x-tile and enc-tile, 2 K-chunks of 32
    const float* xr = x   + (rowbase + ln) * 64;
    const float* er = enc + (rowbase + ln) * 64;
    v16h ax[2], ae[2];
    #pragma unroll
    for (int kc = 0; kc < 2; ++kc) {
        v8f xlo = *(const v8f*)(xr + kc * 32 + 16 * lh);
        v8f xhi = *(const v8f*)(xr + kc * 32 + 16 * lh + 8);
        v8f elo = *(const v8f*)(er + kc * 32 + 16 * lh);
        v8f ehi = *(const v8f*)(er + kc * 32 + 16 * lh + 8);
        ax[kc] = pack16(__builtin_convertvector(xlo, v8h), __builtin_convertvector(xhi, v8h));
        ae[kc] = pack16(__builtin_convertvector(elo, v8h), __builtin_convertvector(ehi, v8h));
    }

    v8f accq = {}, acck = {}, accv = {};
    #pragma unroll
    for (int kc = 0; kc < 2; ++kc) {
        v16h bfq = *(const v16h*)(WqT  + (w * 16 + ln) * 64        + kc * 32 + 16 * lh);
        v16h bfk = *(const v16h*)(WkvT + (w * 16 + ln) * 64        + kc * 32 + 16 * lh);
        v16h bfv = *(const v16h*)(WkvT + (64 + w * 16 + ln) * 64   + kc * 32 + 16 * lh);
        accq = wmma_f16(ax[kc], bfq, accq);
        acck = wmma_f16(ae[kc], bfk, acck);
        accv = wmma_f16(ae[kc], bfv, accv);
    }

    const int col = w * 16 + ln;                  // global output column 0..63
    const int h = w, d = ln;                      // head, head-dim
    const float bQ = bq[col], bK = bkv[col], bV = bkv[64 + col];

    _Float16* Qp = Qh + ((b * 4 + h) * 1024 + t0) * 16 + d;
    _Float16* Kp = Kh + ((b * 4 + h) * 1024 + t0) * 16 + d;
    _Float16* Vp = Vt + ((b * 4 + h) * 16 + d) * 1024 + t0;
    #pragma unroll
    for (int r = 0; r < 8; ++r) {
        const int M = r + 8 * lh;                 // row within tile
        Qp[M * 16] = (_Float16)((accq[r] + bQ) * 0.25f);  // fold 1/sqrt(HD)
        Kp[M * 16] = (_Float16)(acck[r] + bK);
        Vp[M]      = (_Float16)(accv[r] + bV);            // transposed layout
    }
}

// ---------------------------------------------------------------------------
// Kernel 2: attention per (b, h, 16-row q-tile).  grid = 16*4*64 = 4096,
// block = 128 (4 waves); each wave owns a 256-wide slice of T_enc.
//   Phase 1: scores = Q K^T (WMMA, K-dim zero-padded 16->32 in A only; B
//            lanes 16..31 multiply zero A halves, so they load duplicates
//            unconditionally). Mask folded here: masked columns store -inf.
//   Phase 2: branchless softmax, v2h-vectorized LDS scans, wave reductions;
//            P = exp(S - max) overwrites S in place as f16.
//   Phase 3: ctx = P V (WMMA over K=32 chunks), cross-wave combine in LDS.
// ---------------------------------------------------------------------------
__global__ void attn(const _Float16* __restrict__ Qh, const _Float16* __restrict__ Kh,
                     const _Float16* __restrict__ Vt, const int* __restrict__ mask,
                     _Float16* __restrict__ Ctx)
{
    __shared__ _Float16 Sbuf[16 * 1024];   // 32 KB: scores, then P=exp(S-max)
    __shared__ float    ctxp[4 * 256];     // per-wave partial ctx tiles
    __shared__ float    rinv[16];          // 1 / row sums

    const int bid = blockIdx.x;
    const int b = bid >> 8, h = (bid >> 6) & 3, qt = bid & 63;
    const int tid = threadIdx.x, w = tid >> 5, l = tid & 31;
    const int lh = l >> 4, ln = l & 15;

    const _Float16* Qb = Qh + ((b * 4 + h) * 1024 + qt * 16) * 16;
    const _Float16* Kb = Kh + ((b * 4 + h) * 1024) * 16;
    const _Float16* Vb = Vt + ((b * 4 + h) * 16) * 1024;
    const int* mrow = mask + b * 1024;

    // Q A-fragment: rows M=ln, K 0..15 real, 16..31 zero
    v8h z8 = {};
    v16h aq = lh ? pack16(z8, z8) : *(const v16h*)(Qb + ln * 16);

    // ---- Phase 1: score tiles, mask folded into the store ----
    const float NEG = -__builtin_huge_valf();
    for (int tt = w * 16; tt < w * 16 + 16; ++tt) {
        const int tbase = tt * 16;
        v16h bf = *(const v16h*)(Kb + (tbase + ln) * 16);
        __builtin_prefetch(Kb + (tbase + 16 + ln) * 16, 0, 0);
        const bool keep = (mrow[tbase + ln] != 0);   // this lane's column
        v8f c0 = {};
        v8f dfr = wmma_f16(aq, bf, c0);
        #pragma unroll
        for (int r = 0; r < 8; ++r) {
            const float sv = keep ? dfr[r] : NEG;
            Sbuf[(r + 8 * lh) * 1024 + tbase + ln] = (_Float16)sv;
        }
    }
    __syncthreads();

    // ---- Phase 2: softmax; wave w owns rows 4w..4w+3 (branchless, v2h) ----
    for (int rr = 0; rr < 4; ++rr) {
        const int row = w * 4 + rr;
        _Float16* Srow = &Sbuf[row * 1024];
        float mx = NEG;
        for (int i = 2 * l; i < 1024; i += 64) {
            v2h p = *(const v2h*)(Srow + i);
            mx = fmaxf(mx, fmaxf((float)p[0], (float)p[1]));
        }
        mx = wave_max(mx);
        float ssum = 0.f;
        for (int i = 2 * l; i < 1024; i += 64) {
            v2h p = *(const v2h*)(Srow + i);
            const float e0 = __expf((float)p[0] - mx);   // exp(-inf-mx)=0
            const float e1 = __expf((float)p[1] - mx);
            v2h q; q[0] = (_Float16)e0; q[1] = (_Float16)e1;
            *(v2h*)(Srow + i) = q;                       // P overwrites S
            ssum += e0 + e1;
        }
        ssum = wave_sum(ssum);
        if (l == 0) rinv[row] = ssum > 0.f ? 1.f / ssum : 0.f;
    }
    __syncthreads();

    // ---- Phase 3: ctx = P @ V over this wave's slice (8 chunks of K=32) ----
    v8f acc = {};
    for (int c = w * 8; c < w * 8 + 8; ++c) {
        const int tbase = c * 32;
        v16h af = pack16(*(const v8h*)(&Sbuf[ln * 1024 + tbase + 16 * lh]),
                         *(const v8h*)(&Sbuf[ln * 1024 + tbase + 16 * lh + 8]));
        v16h bf = *(const v16h*)(Vb + ln * 1024 + tbase + 16 * lh);
        __builtin_prefetch(Vb + ln * 1024 + tbase + 32 + 16 * lh, 0, 0);
        acc = wmma_f16(af, bf, acc);
    }
    #pragma unroll
    for (int r = 0; r < 8; ++r)
        ctxp[w * 256 + (r + 8 * lh) * 16 + ln] = acc[r];
    __syncthreads();

    // combine 4 wave partials, scale by 1/rowsum, emit f16 ctx [b][t][h*16+n]
    for (int e = tid; e < 256; e += 128) {
        const int m = e >> 4, n = e & 15;
        float v = ctxp[e] + ctxp[256 + e] + ctxp[512 + e] + ctxp[768 + e];
        v *= rinv[m];
        Ctx[((b * 1024 + qt * 16 + m) * 64) + h * 16 + n] = (_Float16)v;
    }
}

// ---------------------------------------------------------------------------
// Kernel 3: out = ctx @ Wproj + bproj.  grid = 16*64 = 1024, block = 128
// (4 waves, wave w computes output column tile w). 2 WMMAs per wave (K=64).
// ---------------------------------------------------------------------------
__global__ void proj(const _Float16* __restrict__ Ctx, const _Float16* __restrict__ WT,
                     const float* __restrict__ bproj, float* __restrict__ out)
{
    const int bid = blockIdx.x;
    const int b = bid >> 6, qt = bid & 63;
    const int tid = threadIdx.x, nt = tid >> 5, l = tid & 31;
    const int lh = l >> 4, ln = l & 15;

    const _Float16* Cb = Ctx + (b * 1024 + qt * 16) * 64;
    v8f acc = {};
    #pragma unroll
    for (int kc = 0; kc < 2; ++kc) {
        v16h af = pack16(*(const v8h*)(Cb + ln * 64 + kc * 32 + 16 * lh),
                         *(const v8h*)(Cb + ln * 64 + kc * 32 + 16 * lh + 8));
        v16h bf = *(const v16h*)(WT + (nt * 16 + ln) * 64 + kc * 32 + 16 * lh);
        acc = wmma_f16(af, bf, acc);
    }
    const int col = nt * 16 + ln;
    const float bias = bproj[col];
    float* ob = out + (b * 1024 + qt * 16) * 64;
    #pragma unroll
    for (int r = 0; r < 8; ++r)
        ob[(r + 8 * lh) * 64 + col] = acc[r] + bias;
}

// ---------------------------------------------------------------------------
// Launch. Workspace (f16): Qh 2MB | Kh 2MB | Vt 2MB | Ctx 2MB | WpT 8KB |
//                          WqT 8KB | WkvT 16KB   (total ~8.4 MB of d_ws)
// ---------------------------------------------------------------------------
extern "C" void kernel_launch(void* const* d_in, const int* in_sizes, int n_in,
                              void* d_out, int out_size, void* d_ws, size_t ws_size,
                              hipStream_t stream)
{
    const float* x    = (const float*)d_in[0];
    const float* enc  = (const float*)d_in[1];
    const int*   msk  = (const int*)  d_in[2];
    const float* Wkv  = (const float*)d_in[3];
    const float* bkv  = (const float*)d_in[4];
    const float* Wq   = (const float*)d_in[5];
    const float* bq   = (const float*)d_in[6];
    const float* Wp   = (const float*)d_in[7];
    const float* bp   = (const float*)d_in[8];

    char* ws = (char*)d_ws;
    _Float16* Qh   = (_Float16*)(ws);
    _Float16* Kh   = (_Float16*)(ws + 1ull * (1u << 21));
    _Float16* Vt   = (_Float16*)(ws + 2ull * (1u << 21));
    _Float16* Ctx  = (_Float16*)(ws + 3ull * (1u << 21));
    _Float16* WpT  = (_Float16*)(ws + 4ull * (1u << 21));
    _Float16* WqT  = (_Float16*)(ws + 4ull * (1u << 21) + 8192);
    _Float16* WkvT = (_Float16*)(ws + 4ull * (1u << 21) + 16384);

    prep_w  <<<256,  64, 0, stream>>>(Wq, Wkv, Wp, WqT, WkvT, WpT);
    prep_qkv<<<1024, 128, 0, stream>>>(x, enc, WqT, WkvT, bkv, bq, Qh, Kh, Vt);
    attn    <<<4096, 128, 0, stream>>>(Qh, Kh, Vt, msk, Ctx);
    proj    <<<1024, 128, 0, stream>>>(Ctx, WpT, bp, (float*)d_out);
}
